// ViT_17025250361594
// MI455X (gfx1250) — matrix operand order repaired and checked by reading, not compiled, not run on hardware
//
#include <hip/hip_runtime.h>
#include <hip/hip_bf16.h>
#include <stdint.h>

// Problem constants (from reference)
#define BB  32
#define SSQ 197
#define EEV 768
#define HHN 12
#define DHH 64
#define FFN 2048
#define LLN 4
#define CCN 1000
#define SPAD 224   // attention row length padded to multiple of 32 (16B aligned)

typedef __attribute__((ext_vector_type(16))) __bf16        v16bf;
typedef __attribute__((ext_vector_type(8)))  float         v8f;
typedef __attribute__((ext_vector_type(4)))  unsigned int  v4u;

__device__ __forceinline__ unsigned short f2b(float f) {
  unsigned int u = __float_as_uint(f);
  u += 0x7FFFu + ((u >> 16) & 1u);          // round-to-nearest-even
  return (unsigned short)(u >> 16);
}
__device__ __forceinline__ float b2f(unsigned short h) {
  return __uint_as_float(((unsigned int)h) << 16);
}
// Generic pointer to LDS byte offset (LDS aperture: addr[31:0] == LDS offset)
__device__ __forceinline__ unsigned lds_off(const void* p) {
  return (unsigned)(uintptr_t)p;
}
// Async DMA: 16 bytes/lane global -> LDS, tracked by ASYNCcnt (gfx1250)
__device__ __forceinline__ void async_g2l_b128(unsigned ldsaddr, const void* gaddr) {
  asm volatile("global_load_async_to_lds_b128 %0, %1, off"
               :: "v"(ldsaddr), "v"(gaddr) : "memory");
}
template<int N> __device__ __forceinline__ void wait_async() {
  if      (N == 0) asm volatile("s_wait_asynccnt 0" ::: "memory");
  else if (N == 2) asm volatile("s_wait_asynccnt 2" ::: "memory");
  else             asm volatile("s_wait_asynccnt 3" ::: "memory");
}

// ---------------------------------------------------------------------------
// Batched WMMA GEMM:  C = act(alpha * A(bf16) x B(bf16) + bias) [+ res(f32)]
//   A : M x K row-major (lda); rows 16B aligned; K multiple of 32.
//   B : TRANSB ? N x K row-major (ldb = K stride, rows 16B aligned)
//             : K x N row-major (ldb = N stride); KB = valid K rows of B.
//   C : f32 or bf16; batched via z = b*Hd + h with per-operand strides.
// Tile 128x64 per 256-thread block; 8 wave32s in 4(M) x 2(N) grid; each wave
// runs a 2x2 grid of v_wmma_f32_16x16x32_bf16 fragments.
// Double-buffered LDS; staging via GLOBAL_LOAD_ASYNC_TO_LDS_B128 pipelined
// one K-tile ahead (ASYNCcnt in-order completion => wait<=C frees tile k).
// OOB rows/cols are index-CLAMPED, not masked: they only feed C entries the
// epilogue never stores, so staging is branch-free.
// ---------------------------------------------------------------------------
template<int ACT, int TRANSB, int OUT_BF16, int HAS_BIAS, int HAS_RES>
__global__ void __launch_bounds__(256)
gemm_wmma(const unsigned short* __restrict__ A, int lda, long sAb, long sAh,
          const unsigned short* __restrict__ Bm, int ldb, long sBb, long sBh,
          const float* __restrict__ bias,
          const float* __restrict__ res,
          void* __restrict__ Cout, int ldc, long sCb, long sCh,
          int M, int N, int K, int KB, int Hd, float alpha)
{
  __shared__ alignas(16) unsigned short As[2][128][40];  // +8 pad, rows 16B aligned
  __shared__ alignas(16) unsigned short Bs[2][64][40];   // stored [N][K]
  const unsigned ABUF = 128 * 40 * 2;   // bytes between A buffers
  const unsigned BBUF = 64 * 40 * 2;    // bytes between B buffers

  const int tid  = threadIdx.x;
  const int lane = tid & 31;
  const int wave = tid >> 5;
  const int wm   = wave & 3;    // M quadrant (32 rows)
  const int wn   = wave >> 2;   // N half (32 cols)

  const int tileM = blockIdx.y * 128;
  const int tileN = blockIdx.x * 64;
  const int zb = blockIdx.z / Hd;
  const int zh = blockIdx.z % Hd;
  A  += (long)zb * sAb + (long)zh * sAh;
  Bm += (long)zb * sBb + (long)zh * sBh;
  const long coff = (long)zb * sCb + (long)zh * sCh;

  v8f acc[2][2];
#pragma unroll
  for (int i = 0; i < 2; i++)
#pragma unroll
    for (int j = 0; j < 2; j++) acc[i][j] = (v8f)(0.0f);

  // ---- loop-invariant staging descriptors (index-clamped, branch-free) ----
  const int c8 = (tid & 3) << 3;                  // 8-element chunk column
  const int r0 = tid >> 2, r1 = r0 + 64;          // A rows (2 chunks/thread)
  int gm0 = tileM + r0; if (gm0 >= M) gm0 = M - 1;
  int gm1 = tileM + r1; if (gm1 >= M) gm1 = M - 1;
  const unsigned short* pA0 = A + (long)gm0 * lda + c8;
  const unsigned short* pA1 = A + (long)gm1 * lda + c8;
  const unsigned lA0 = lds_off(&As[0][r0][c8]);
  const unsigned lA1 = lds_off(&As[0][r1][c8]);
  const int nB = tid >> 2;                        // B row (TRANSB, 1 chunk/thread)
  int gnB = tileN + nB; if (gnB >= N) gnB = N - 1;
  const unsigned short* pB = Bm + (long)gnB * ldb + c8;
  const unsigned lB = lds_off(&Bs[0][nB][c8]);

  int ks = 0;   // K position of next tile to stage
  auto stage = [&](int buf) {
    const unsigned ao = buf ? ABUF : 0u;
    async_g2l_b128(lA0 + ao, pA0); pA0 += 32;
    async_g2l_b128(lA1 + ao, pA1); pA1 += 32;
    if (TRANSB) {
      async_g2l_b128(lB + (buf ? BBUF : 0u), pB); pB += 32;
    } else {
      // K x N memory: vector load along N, scatter-transpose into LDS.
      // K rows >= KB are clamped: they pair with zero-padded A columns.
      const int kk = tid >> 3;               // 0..31
      const int n8 = (tid & 7) << 3;         // 0..56
      int gk = ks + kk; if (gk >= KB) gk = KB - 1;
      v4u val = (v4u)(0u);
      if ((tileN + n8 + 8) <= N)
        val = *(const v4u*)(const void*)(Bm + (long)gk * ldb + (tileN + n8));
      const unsigned short* sv = (const unsigned short*)&val;
#pragma unroll
      for (int j = 0; j < 8; j++) Bs[buf][n8 + j][kk] = sv[j];
    }
    ks += 32;
  };

  // ISA 7.12.2 fragment lane mapping (16-bit, 16x16x32)
  const int arow = lane & 15;
  const int asel = (lane >> 4) * 8;      // lanes 0-15: K 0..7/16..23; 16-31: 8..15/24..31
  const int bcol = lane & 15;
  const int bsel = (lane >> 4) * 8;      // lanes 0-15: K 0..7/16..23; 16-31: 8..15/24..31

  const int nkt = K >> 5;                // K is a multiple of 32
  stage(0);                              // prologue: tile 0 -> buffer 0

  for (int kt = 0; kt < nkt; kt++) {
    const int cur = kt & 1;
    if (kt + 1 < nkt) {
      stage(cur ^ 1);                    // issue tile kt+1 into other buffer
      wait_async<TRANSB ? 3 : 2>();      // in-order => tile kt has landed
    } else {
      wait_async<0>();
    }
    __syncthreads();                     // all waves' tiles visible

    v16bf afrag[2], bfrag[2];
#pragma unroll
    for (int f = 0; f < 2; f++) {
      union { v4u u[2]; v16bf v; } ua;
      const unsigned short* p = &As[cur][wm * 32 + f * 16 + arow][asel];
      ua.u[0] = *(const v4u*)(const void*)p;
      ua.u[1] = *(const v4u*)(const void*)(p + 16);
      afrag[f] = ua.v;
    }
#pragma unroll
    for (int f = 0; f < 2; f++) {
      union { v4u u[2]; v16bf v; } ub;
      const unsigned short* p = &Bs[cur][wn * 32 + f * 16 + bcol][bsel];
      ub.u[0] = *(const v4u*)(const void*)p;
      ub.u[1] = *(const v4u*)(const void*)(p + 16);
      bfrag[f] = ub.v;
    }
#pragma unroll
    for (int i = 0; i < 2; i++)
#pragma unroll
      for (int j = 0; j < 2; j++)
        acc[i][j] = __builtin_amdgcn_wmma_f32_16x16x32_bf16(
            false, afrag[i], false, bfrag[j], (short)0, acc[i][j], false, false);
    __syncthreads();   // reads done before buffer is re-staged two tiles later
  }

  // ---- epilogue: alpha, bias, GELU, residual, store ----
  const int cn   = lane & 15;
  const int rsel = (lane >> 4) * 8;
#pragma unroll
  for (int fm = 0; fm < 2; fm++)
#pragma unroll
    for (int fn = 0; fn < 2; fn++) {
      int gn = tileN + wn * 32 + fn * 16 + cn;
      if (gn >= N) continue;
#pragma unroll
      for (int r = 0; r < 8; r++) {
        int gm = tileM + wm * 32 + fm * 16 + rsel + r;
        if (gm >= M) continue;
        float v = acc[fm][fn][r] * alpha;
        if (HAS_BIAS) v += bias[gn];
        if (ACT == 1) v = 0.5f * v * (1.0f + erff(v * 0.70710678f)); // exact GELU
        long ci = coff + (long)gm * ldc + gn;
        if (HAS_RES) v += res[ci];
        if (OUT_BF16) ((unsigned short*)Cout)[ci] = f2b(v);
        else          ((float*)Cout)[ci] = v;
      }
    }
}

// ---------------------------------------------------------------------------
// LayerNorm over E=768, one block per row, bf16 output
// ---------------------------------------------------------------------------
__global__ void __launch_bounds__(256)
layernorm_kernel(const float* __restrict__ x, const float* __restrict__ w,
                 const float* __restrict__ b, unsigned short* __restrict__ out)
{
  const int row = blockIdx.x, tid = threadIdx.x;
  const float* xr = x + (long)row * EEV;
  __shared__ float red[256];
  float s = 0.f, s2 = 0.f;
  for (int e = tid; e < EEV; e += 256) { float v = xr[e]; s += v; s2 += v * v; }
  red[tid] = s; __syncthreads();
  for (int st = 128; st > 0; st >>= 1) { if (tid < st) red[tid] += red[tid + st]; __syncthreads(); }
  float mean = red[0] * (1.0f / EEV); __syncthreads();
  red[tid] = s2; __syncthreads();
  for (int st = 128; st > 0; st >>= 1) { if (tid < st) red[tid] += red[tid + st]; __syncthreads(); }
  float var  = red[0] * (1.0f / EEV) - mean * mean;
  float rstd = rsqrtf(var + 1e-5f);
  for (int e = tid; e < EEV; e += 256)
    out[(long)row * EEV + e] = f2b((xr[e] - mean) * rstd * w[e] + b[e]);
}

// ---------------------------------------------------------------------------
// Row softmax over S=197 bf16 scores (ld = SPAD, zero-fills padding columns)
// ---------------------------------------------------------------------------
__global__ void __launch_bounds__(256)
softmax_att_kernel(unsigned short* __restrict__ att)
{
  const long row = blockIdx.x;
  const int tid = threadIdx.x;
  unsigned short* ar = att + row * SPAD;
  __shared__ float red[256];
  float x = 0.f, v = -3.0e38f;
  if (tid < SSQ) { x = b2f(ar[tid]); v = x; }
  red[tid] = v; __syncthreads();
  for (int st = 128; st > 0; st >>= 1) { if (tid < st) red[tid] = fmaxf(red[tid], red[tid + st]); __syncthreads(); }
  float m = red[0]; __syncthreads();
  float e = (tid < SSQ) ? __expf(x - m) : 0.f;
  red[tid] = e; __syncthreads();
  for (int st = 128; st > 0; st >>= 1) { if (tid < st) red[tid] += red[tid + st]; __syncthreads(); }
  float inv = 1.0f / red[0];
  if (tid < SSQ)            ar[tid] = f2b(e * inv);
  else if (tid < SPAD)      ar[tid] = 0;           // zero pad -> K=224 GEMM is exact
}

// ---------------------------------------------------------------------------
// Classifier + output softmax (49 MFLOP, scalar path is fine)
// ---------------------------------------------------------------------------
__global__ void __launch_bounds__(256)
classifier_kernel(const float* __restrict__ h, const float* __restrict__ Wc,
                  const float* __restrict__ bc, float* __restrict__ logits)
{
  const int b = blockIdx.y;
  const int c = blockIdx.x * 256 + threadIdx.x;
  if (c >= CCN) return;
  const float* hr = h + (long)b * SSQ * EEV;  // token 0 of batch b
  float acc = bc[c];
  for (int e = 0; e < EEV; e++) acc = fmaf(hr[e], Wc[(long)e * CCN + c], acc);
  logits[(long)b * CCN + c] = acc;
}

__global__ void __launch_bounds__(256)
softmax_out_kernel(const float* __restrict__ logits, float* __restrict__ out)
{
  const int b = blockIdx.x, tid = threadIdx.x;
  const float* lr = logits + (long)b * CCN;
  __shared__ float red[256];
  float m = -3.0e38f;
  for (int c = tid; c < CCN; c += 256) m = fmaxf(m, lr[c]);
  red[tid] = m; __syncthreads();
  for (int st = 128; st > 0; st >>= 1) { if (tid < st) red[tid] = fmaxf(red[tid], red[tid + st]); __syncthreads(); }
  m = red[0]; __syncthreads();
  float s = 0.f;
  for (int c = tid; c < CCN; c += 256) s += __expf(lr[c] - m);
  red[tid] = s; __syncthreads();
  for (int st = 128; st > 0; st >>= 1) { if (tid < st) red[tid] += red[tid + st]; __syncthreads(); }
  float inv = 1.0f / red[0];
  for (int c = tid; c < CCN; c += 256) out[(long)b * CCN + c] = __expf(lr[c] - m) * inv;
}

// ---------------------------------------------------------------------------
// Weight prep: fp32 -> bf16, pre-TRANSPOSED to [N][K] for b128 staging.
// QKV: (L,H,E,D) -> dst[l][h*D+d][e]
// ---------------------------------------------------------------------------
__global__ void __launch_bounds__(256)
pack_qkvT_kernel(const float* __restrict__ src, unsigned short* __restrict__ dst)
{
  long i = (long)blockIdx.x * 256 + threadIdx.x;
  const long n = (long)LLN * HHN * EEV * DHH;
  if (i >= n) return;
  int d = (int)(i % DHH);
  long t = i / DHH;
  int e = (int)(t % EEV); t /= EEV;
  int h = (int)(t % HHN); t /= HHN;
  int l = (int)t;
  dst[((long)l * (HHN * DHH) + h * DHH + d) * EEV + e] = f2b(src[i]);
}

// Generic (L, K, N) row-major -> (L, N, K) bf16 transpose
__global__ void __launch_bounds__(256)
cvtT_kernel(const float* __restrict__ src, unsigned short* __restrict__ dst,
            int K, int N, long total)
{
  long i = (long)blockIdx.x * 256 + threadIdx.x;
  if (i >= total) return;
  long kn = (long)K * N;
  int  l  = (int)(i / kn);
  long r  = i % kn;
  int  n  = (int)(r / K);
  int  k  = (int)(r % K);
  dst[i] = f2b(src[(long)l * kn + (long)k * N + n]);
}

// ---------------------------------------------------------------------------
extern "C" void kernel_launch(void* const* d_in, const int* in_sizes, int n_in,
                              void* d_out, int out_size, void* d_ws, size_t ws_size,
                              hipStream_t stream)
{
  (void)in_sizes; (void)n_in; (void)out_size; (void)ws_size;
  const float* x     = (const float*)d_in[0];
  const float* Wk    = (const float*)d_in[1];
  const float* Wq    = (const float*)d_in[2];
  const float* Wv    = (const float*)d_in[3];
  const float* Wconv = (const float*)d_in[4];
  const float* ln1w  = (const float*)d_in[5];
  const float* ln1b  = (const float*)d_in[6];
  const float* ln2w  = (const float*)d_in[7];
  const float* ln2b  = (const float*)d_in[8];
  const float* W1    = (const float*)d_in[9];
  const float* b1    = (const float*)d_in[10];
  const float* W2    = (const float*)d_in[11];
  const float* b2    = (const float*)d_in[12];
  const float* Wc    = (const float*)d_in[13];
  const float* bc    = (const float*)d_in[14];

  const int M = BB * SSQ;  // 6304 token rows

  // --- workspace carve ---
  char* ws = (char*)d_ws;
  size_t off = 0;
  auto carve = [&](size_t bytes) -> void* {
    void* p = ws + off;
    off += (bytes + 255) & ~(size_t)255;
    return p;
  };
  unsigned short* wqT    = (unsigned short*)carve((size_t)LLN * EEV * 768 * 2);
  unsigned short* wkT    = (unsigned short*)carve((size_t)LLN * EEV * 768 * 2);
  unsigned short* wvT    = (unsigned short*)carve((size_t)LLN * EEV * 768 * 2);
  unsigned short* wcvT   = (unsigned short*)carve((size_t)LLN * 768 * EEV * 2);
  unsigned short* w1T    = (unsigned short*)carve((size_t)LLN * EEV * FFN * 2);
  unsigned short* w2T    = (unsigned short*)carve((size_t)LLN * FFN * EEV * 2);
  unsigned short* xnb    = (unsigned short*)carve((size_t)M * EEV * 2);
  unsigned short* qb     = (unsigned short*)carve((size_t)M * 768 * 2);
  unsigned short* kb     = (unsigned short*)carve((size_t)M * 768 * 2);
  unsigned short* vb     = (unsigned short*)carve((size_t)M * 768 * 2);
  unsigned short* attb   = (unsigned short*)carve((size_t)BB * HHN * SSQ * SPAD * 2);
  unsigned short* ob     = (unsigned short*)carve((size_t)M * 768 * 2);
  unsigned short* ff1b   = (unsigned short*)carve((size_t)M * FFN * 2);
  float*          hbuf   = (float*)carve((size_t)M * EEV * 4);
  float*          res1   = (float*)carve((size_t)M * EEV * 4);
  float*          logitsf= (float*)carve((size_t)BB * CCN * 4);

  // --- weight prep (bf16, transposed to [N][K]) ---
  const long nW = (long)LLN * HHN * EEV * DHH;          // == L*768*768
  const int  gW = (int)((nW + 255) / 256);
  pack_qkvT_kernel<<<gW, 256, 0, stream>>>(Wq, wqT);
  pack_qkvT_kernel<<<gW, 256, 0, stream>>>(Wk, wkT);
  pack_qkvT_kernel<<<gW, 256, 0, stream>>>(Wv, wvT);
  cvtT_kernel<<<gW, 256, 0, stream>>>(Wconv, wcvT, 768, EEV, nW);
  const long nF = (long)LLN * EEV * FFN;
  const int  gF = (int)((nF + 255) / 256);
  cvtT_kernel<<<gF, 256, 0, stream>>>(W1, w1T, EEV, FFN, nF);
  cvtT_kernel<<<gF, 256, 0, stream>>>(W2, w2T, FFN, EEV, nF);

  hipMemcpyAsync(hbuf, x, (size_t)M * EEV * sizeof(float),
                 hipMemcpyDeviceToDevice, stream);

  const float inv_scale = 0.071247046f;  // 1/sqrt(197) — ref scales by sqrt(seq_len)
  const dim3 g768(12, 50, 1);            // N=768, M=6304
  const dim3 gF1(32, 50, 1);             // N=2048
  const dim3 gSc(4, 2, BB * HHN);        // scores: 197x197 per (b,h)
  const dim3 gAV(1, 2, BB * HHN);        // attn*V: 197x64 per (b,h)
  const long sTok = (long)SSQ * 768;     // per-batch token-row stride (elements)
  const long sAtt = (long)SSQ * SPAD;    // per-head score-tile stride

  for (int l = 0; l < LLN; l++) {
    const unsigned short* wq  = wqT  + (long)l * EEV * 768;
    const unsigned short* wk  = wkT  + (long)l * EEV * 768;
    const unsigned short* wv  = wvT  + (long)l * EEV * 768;
    const unsigned short* wcv = wcvT + (long)l * 768 * EEV;
    const unsigned short* w1p = w1T  + (long)l * EEV * FFN;
    const unsigned short* w2p = w2T  + (long)l * FFN * EEV;

    // xn = LN1(h)
    layernorm_kernel<<<M, 256, 0, stream>>>(hbuf, ln1w + l * EEV, ln1b + l * EEV, xnb);
    // q/k/v = xn @ W^T   (TRANSB weights, bf16 out, layout [b*S, h*64+d])
    gemm_wmma<0,1,1,0,0><<<g768, 256, 0, stream>>>(xnb, EEV, 0, 0, wq, EEV, 0, 0,
        nullptr, nullptr, qb, 768, 0, 0, M, 768, EEV, EEV, 1, 1.0f);
    gemm_wmma<0,1,1,0,0><<<g768, 256, 0, stream>>>(xnb, EEV, 0, 0, wk, EEV, 0, 0,
        nullptr, nullptr, kb, 768, 0, 0, M, 768, EEV, EEV, 1, 1.0f);
    gemm_wmma<0,1,1,0,0><<<g768, 256, 0, stream>>>(xnb, EEV, 0, 0, wv, EEV, 0, 0,
        nullptr, nullptr, vb, 768, 0, 0, M, 768, EEV, EEV, 1, 1.0f);
    // scores = (q . k^T) * inv_scale  into padded [197 x 224] tiles
    gemm_wmma<0,1,1,0,0><<<gSc, 256, 0, stream>>>(qb, 768, sTok, 64, kb, 768, sTok, 64,
        nullptr, nullptr, attb, SPAD, (long)HHN * sAtt, sAtt,
        SSQ, SSQ, 64, 64, HHN, inv_scale);
    // softmax rows (writes zeros into pad columns)
    softmax_att_kernel<<<BB * HHN * SSQ, 256, 0, stream>>>(attb);
    // o = att @ v : K padded to 224 (pad columns are zero), KB bounds V reads
    gemm_wmma<0,0,1,0,0><<<gAV, 256, 0, stream>>>(attb, SPAD, (long)HHN * sAtt, sAtt,
        vb, 768, sTok, 64, nullptr, nullptr, ob, 768, sTok, 64,
        SSQ, DHH, SPAD, SSQ, HHN, 1.0f);
    // res1 = o @ Wconv + h
    gemm_wmma<0,1,0,0,1><<<g768, 256, 0, stream>>>(ob, 768, 0, 0, wcv, 768, 0, 0,
        nullptr, hbuf, res1, 768, 0, 0, M, 768, 768, 768, 1, 1.0f);
    // xn2 = LN2(res1)
    layernorm_kernel<<<M, 256, 0, stream>>>(res1, ln2w + l * EEV, ln2b + l * EEV, xnb);
    // ff1 = gelu(xn2 @ W1 + b1)
    gemm_wmma<1,1,1,1,0><<<gF1, 256, 0, stream>>>(xnb, EEV, 0, 0, w1p, EEV, 0, 0,
        b1 + (long)l * FFN, nullptr, ff1b, FFN, 0, 0, M, FFN, EEV, EEV, 1, 1.0f);
    // h = ff1 @ W2 + b2 + res1
    gemm_wmma<0,1,0,1,1><<<g768, 256, 0, stream>>>(ff1b, FFN, 0, 0, w2p, FFN, 0, 0,
        b2 + (long)l * EEV, res1, hbuf, EEV, 0, 0, M, EEV, FFN, FFN, 1, 1.0f);
  }

  classifier_kernel<<<dim3((CCN + 255) / 256, BB), 256, 0, stream>>>(hbuf, Wc, bc, logitsf);
  softmax_out_kernel<<<BB, 256, 0, stream>>>(logitsf, (float*)d_out);
}
